// DSNaiveMoE_52733608460326
// MI455X (gfx1250) — matrix operand order repaired and checked by reading, not compiled, not run on hardware
//
#include <hip/hip_runtime.h>

#define E_EXP 8
#define DIMK 1024
#define HH 704
#define TWO_H 1408
#define T_TOK 4096
#define NSLOT 8192           // T*TOPK
#define MROWS 32
#define NT1 88               // 1408/16
#define KS1 32               // 1024/32
#define NT2 64               // 1024/16
#define KS2 22               // 704/32

typedef __bf16 bf16_t;
typedef bf16_t v16bf __attribute__((ext_vector_type(16)));
typedef float v8f __attribute__((ext_vector_type(8)));

static __device__ __forceinline__ unsigned short f2b(float f) {
  unsigned int u = __float_as_uint(f);
  u += 0x7fffu + ((u >> 16) & 1u);          // round-to-nearest-even to bf16
  return (unsigned short)(u >> 16);
}
static __device__ __forceinline__ float b2f(unsigned short h) {
  return __uint_as_float(((unsigned int)h) << 16);
}

// LDS A-fragment: 16-bit A 16x32 layout -> slots 0-7 = k..k+7, slots 8-15 = k+16..k+23
static __device__ __forceinline__ v16bf frag_lds(const unsigned short* p) {
  union { uint4 q[2]; v16bf v; } u;
  u.q[0] = *reinterpret_cast<const uint4*>(p);
  u.q[1] = *reinterpret_cast<const uint4*>(p + 16);
  return u.v;
}
// Pre-swizzled global B-fragment: 16 contiguous u16 per lane
static __device__ __forceinline__ v16bf frag_glb(const unsigned short* p) {
  union { uint4 q[2]; v16bf v; } u;
  u.q[0] = *reinterpret_cast<const uint4*>(p);
  u.q[1] = *reinterpret_cast<const uint4*>(p + 8);
  return u.v;
}

#define WMMA_BF16(a, b, c) \
  __builtin_amdgcn_wmma_f32_16x16x32_bf16(false, (a), false, (b), (short)0, (c), false, false)

// ---------------- routing ----------------
__global__ void route_init(int* counts, int* cursor) {
  int i = threadIdx.x;
  if (i < E_EXP) { counts[i] = 0; cursor[i] = 0; }
}
__global__ void route_count(const int* __restrict__ idx, int* __restrict__ counts) {
  int i = blockIdx.x * blockDim.x + threadIdx.x;
  if (i < NSLOT) atomicAdd(&counts[idx[i] & (E_EXP - 1)], 1);
}
__global__ void route_scan(const int* __restrict__ counts, int* __restrict__ offsets,
                           int* __restrict__ cursor) {
  if (threadIdx.x == 0 && blockIdx.x == 0) {
    int acc = 0;
    for (int e = 0; e < E_EXP; ++e) { offsets[e] = acc; cursor[e] = acc; acc += counts[e]; }
    offsets[E_EXP] = acc;
  }
}
__global__ void route_scatter(const int* __restrict__ idx, const float* __restrict__ wts,
                              int* __restrict__ cursor, int* __restrict__ slot_token,
                              float* __restrict__ slot_gate) {
  int i = blockIdx.x * blockDim.x + threadIdx.x;
  if (i < NSLOT) {
    int e = idx[i] & (E_EXP - 1);
    int pos = atomicAdd(&cursor[e], 1);
    slot_token[pos] = i >> 1;   // TOPK = 2
    slot_gate[pos]  = wts[i];
  }
}
__global__ void zero_f32(float* __restrict__ p, int n4) {
  int i = blockIdx.x * blockDim.x + threadIdx.x;
  if (i < n4) reinterpret_cast<float4*>(p)[i] = (float4){0.f, 0.f, 0.f, 0.f};
}

// ------------- weight pre-pack (f32 -> bf16 hi/lo, WMMA-frag swizzled) -------------
// frag id = ((e*NT + nt)*KS + kstep); per frag: [lane][s] = W[k = kstep*32 + (lane>>4)*8 + (s>>3)*16 + (s&7)][n = nt*16 + (lane&15)]
__global__ void pack_w12(const float* __restrict__ W12, unsigned short* __restrict__ ph,
                         unsigned short* __restrict__ pl) {
  int gid  = blockIdx.x * 256 + threadIdx.x;   // 5632*256 = 22528 frags * 64
  int a    = gid & 1;
  int lane = (gid >> 1) & 31;
  int frag = gid >> 6;
  int kg   = frag & 31;        // KS1 = 32
  int rest = frag >> 5;
  int nt   = rest % NT1;
  int e    = rest / NT1;
  int n     = nt * 16 + (lane & 15);
  int kbase = kg * 32 + ((lane >> 4) << 3) + (a << 4);
  const float* src = W12 + ((size_t)e * DIMK + kbase) * TWO_H + n;
  unsigned short hs[8], ls[8];
#pragma unroll
  for (int c = 0; c < 8; ++c) {
    float w = src[(size_t)c * TWO_H];
    hs[c] = f2b(w);
    ls[c] = f2b(w - b2f(hs[c]));
  }
  size_t ob = (size_t)frag * 512 + lane * 16 + a * 8;
  uint4 qh, ql;
  qh.x = (unsigned)hs[0] | ((unsigned)hs[1] << 16); qh.y = (unsigned)hs[2] | ((unsigned)hs[3] << 16);
  qh.z = (unsigned)hs[4] | ((unsigned)hs[5] << 16); qh.w = (unsigned)hs[6] | ((unsigned)hs[7] << 16);
  ql.x = (unsigned)ls[0] | ((unsigned)ls[1] << 16); ql.y = (unsigned)ls[2] | ((unsigned)ls[3] << 16);
  ql.z = (unsigned)ls[4] | ((unsigned)ls[5] << 16); ql.w = (unsigned)ls[6] | ((unsigned)ls[7] << 16);
  *reinterpret_cast<uint4*>(ph + ob) = qh;
  *reinterpret_cast<uint4*>(pl + ob) = ql;
}
__global__ void pack_w3(const float* __restrict__ W3, unsigned short* __restrict__ ph,
                        unsigned short* __restrict__ pl) {
  int gid  = blockIdx.x * 256 + threadIdx.x;   // 2816*256 = 11264 frags * 64
  int a    = gid & 1;
  int lane = (gid >> 1) & 31;
  int frag = gid >> 6;
  int kg   = frag % KS2;
  int rest = frag / KS2;
  int nt   = rest & 63;
  int e    = rest >> 6;
  int n     = nt * 16 + (lane & 15);
  int kbase = kg * 32 + ((lane >> 4) << 3) + (a << 4);
  const float* src = W3 + ((size_t)e * HH + kbase) * DIMK + n;
  unsigned short hs[8], ls[8];
#pragma unroll
  for (int c = 0; c < 8; ++c) {
    float w = src[(size_t)c * DIMK];
    hs[c] = f2b(w);
    ls[c] = f2b(w - b2f(hs[c]));
  }
  size_t ob = (size_t)frag * 512 + lane * 16 + a * 8;
  uint4 qh, ql;
  qh.x = (unsigned)hs[0] | ((unsigned)hs[1] << 16); qh.y = (unsigned)hs[2] | ((unsigned)hs[3] << 16);
  qh.z = (unsigned)hs[4] | ((unsigned)hs[5] << 16); qh.w = (unsigned)hs[6] | ((unsigned)hs[7] << 16);
  ql.x = (unsigned)ls[0] | ((unsigned)ls[1] << 16); ql.y = (unsigned)ls[2] | ((unsigned)ls[3] << 16);
  ql.z = (unsigned)ls[4] | ((unsigned)ls[5] << 16); ql.w = (unsigned)ls[6] | ((unsigned)ls[7] << 16);
  *reinterpret_cast<uint4*>(ph + ob) = qh;
  *reinterpret_cast<uint4*>(pl + ob) = ql;
}

// ---------------- main fused MoE kernel ----------------
__global__ __launch_bounds__(256) void moe_main(
    const float* __restrict__ hidden, const float* __restrict__ b12,
    const float* __restrict__ b3, const unsigned short* __restrict__ pw12h,
    const unsigned short* __restrict__ pw12l, const unsigned short* __restrict__ pw3h,
    const unsigned short* __restrict__ pw3l, const int* __restrict__ offsets,
    const int* __restrict__ slot_token, const float* __restrict__ slot_gate,
    float* __restrict__ out) {
  const int e   = blockIdx.y;
  const int beg = offsets[e];
  const int end = offsets[e + 1];
  const int base = beg + blockIdx.x * MROWS;
  if (base >= end) return;
  const int nvalid = min(MROWS, end - base);

  __shared__ unsigned short Ah[32][264];   // A chunk hi (K-block of 256), padded stride
  __shared__ unsigned short Al[32][264];   // A chunk lo
  __shared__ unsigned short Hh[32][712];   // hidden hi (32 x 704)
  __shared__ unsigned short Hl[32][712];   // hidden lo
  __shared__ int   sTok[32];
  __shared__ float sGate[32];

  const int tid   = threadIdx.x;
  const int lane  = tid & 31;
  const int wave  = tid >> 5;
  const int lhalf = lane >> 4;
  const int llow  = lane & 15;

  if (tid < 32) {
    sTok[tid]  = (tid < nvalid) ? slot_token[base + tid] : 0;
    sGate[tid] = (tid < nvalid) ? slot_gate[base + tid] : 0.0f;
  }
  __syncthreads();

  // wave owns column pairs p (x1 tile p <-> x2 tile p+44), p = wave, wave+8, ...
  v8f acc1[6][2][2];
#pragma unroll
  for (int pi = 0; pi < 6; ++pi)
#pragma unroll
    for (int w2 = 0; w2 < 2; ++w2)
#pragma unroll
      for (int m = 0; m < 2; ++m)
        acc1[pi][w2][m] = (v8f){0.f, 0.f, 0.f, 0.f, 0.f, 0.f, 0.f, 0.f};

  for (int kc = 0; kc < 4; ++kc) {          // K blocked by 256
    __syncthreads();
#pragma unroll
    for (int r = 0; r < 8; ++r) {           // stage 32x256 A chunk, split to bf16 hi/lo
      int fidx = r * 256 + tid;
      int row = fidx >> 6;
      int c4  = fidx & 63;
      float4 v = reinterpret_cast<const float4*>(hidden)[(size_t)sTok[row] * 256 + kc * 64 + c4];
      unsigned short h0 = f2b(v.x), h1 = f2b(v.y), h2 = f2b(v.z), h3 = f2b(v.w);
      unsigned short l0 = f2b(v.x - b2f(h0)), l1 = f2b(v.y - b2f(h1));
      unsigned short l2 = f2b(v.z - b2f(h2)), l3 = f2b(v.w - b2f(h3));
      uint2 ph, pl;
      ph.x = (unsigned)h0 | ((unsigned)h1 << 16); ph.y = (unsigned)h2 | ((unsigned)h3 << 16);
      pl.x = (unsigned)l0 | ((unsigned)l1 << 16); pl.y = (unsigned)l2 | ((unsigned)l3 << 16);
      *reinterpret_cast<uint2*>(&Ah[row][c4 * 4]) = ph;
      *reinterpret_cast<uint2*>(&Al[row][c4 * 4]) = pl;
    }
    __syncthreads();

    for (int ks = 0; ks < 8; ++ks) {
      const int kg = kc * 8 + ks;
      v16bf fah[2], fal[2];
#pragma unroll
      for (int m = 0; m < 2; ++m) {
        const unsigned short* pa = &Ah[m * 16 + llow][ks * 32 + (lhalf << 3)];
        const unsigned short* pb = &Al[m * 16 + llow][ks * 32 + (lhalf << 3)];
        fah[m] = frag_lds(pa);
        fal[m] = frag_lds(pb);
      }
#pragma unroll
      for (int pi = 0; pi < 6; ++pi) {
        const int p = wave + 8 * pi;
        if (p < 44) {
#pragma unroll
          for (int w2 = 0; w2 < 2; ++w2) {
            const int nt = p + w2 * 44;
            const size_t fb = ((size_t)((e * NT1 + nt) * KS1 + kg)) * 512 + lane * 16;
            v16bf fbh = frag_glb(pw12h + fb);
            v16bf fbl = frag_glb(pw12l + fb);
#pragma unroll
            for (int m = 0; m < 2; ++m) {
              acc1[pi][w2][m] = WMMA_BF16(fah[m], fbh, acc1[pi][w2][m]);
              acc1[pi][w2][m] = WMMA_BF16(fah[m], fbl, acc1[pi][w2][m]);
              acc1[pi][w2][m] = WMMA_BF16(fal[m], fbh, acc1[pi][w2][m]);
            }
          }
        }
      }
    }
  }

  // bias + SiLU(x1)*x2, write hidden to LDS as bf16 hi/lo
#pragma unroll
  for (int pi = 0; pi < 6; ++pi) {
    const int p = wave + 8 * pi;
    if (p < 44) {
      const float bx1 = b12[e * TWO_H + p * 16 + llow];
      const float bx2 = b12[e * TWO_H + HH + p * 16 + llow];
#pragma unroll
      for (int m = 0; m < 2; ++m) {
#pragma unroll
        for (int i = 0; i < 8; ++i) {
          float x1 = acc1[pi][0][m][i] + bx1;
          float x2 = acc1[pi][1][m][i] + bx2;
          float hv = (x1 / (1.0f + __expf(-x1))) * x2;
          int row = m * 16 + i + (lhalf << 3);
          int col = p * 16 + llow;
          unsigned short hh = f2b(hv);
          unsigned short hl = f2b(hv - b2f(hh));
          Hh[row][col] = hh;
          Hl[row][col] = hl;
        }
      }
    }
  }
  __syncthreads();

  // GEMM2: (32 x 704) @ W3[e] (704 x 1024); wave owns 8 consecutive N tiles
  v8f acc2[8][2];
#pragma unroll
  for (int j = 0; j < 8; ++j)
#pragma unroll
    for (int m = 0; m < 2; ++m)
      acc2[j][m] = (v8f){0.f, 0.f, 0.f, 0.f, 0.f, 0.f, 0.f, 0.f};

  for (int ks = 0; ks < KS2; ++ks) {
    v16bf fah[2], fal[2];
#pragma unroll
    for (int m = 0; m < 2; ++m) {
      const unsigned short* pa = &Hh[m * 16 + llow][ks * 32 + (lhalf << 3)];
      const unsigned short* pb = &Hl[m * 16 + llow][ks * 32 + (lhalf << 3)];
      fah[m] = frag_lds(pa);
      fal[m] = frag_lds(pb);
    }
#pragma unroll
    for (int j = 0; j < 8; ++j) {
      const int nt = wave * 8 + j;
      const size_t fb = ((size_t)((e * NT2 + nt) * KS2 + ks)) * 512 + lane * 16;
      v16bf fbh = frag_glb(pw3h + fb);
      v16bf fbl = frag_glb(pw3l + fb);
#pragma unroll
      for (int m = 0; m < 2; ++m) {
        acc2[j][m] = WMMA_BF16(fah[m], fbh, acc2[j][m]);
        acc2[j][m] = WMMA_BF16(fah[m], fbl, acc2[j][m]);
        acc2[j][m] = WMMA_BF16(fal[m], fbh, acc2[j][m]);
      }
    }
  }

  // epilogue: + b3, scale by gate, atomic accumulate into out[token]
#pragma unroll
  for (int j = 0; j < 8; ++j) {
    const int col = (wave * 8 + j) * 16 + llow;
    const float bv = b3[e * DIMK + col];
#pragma unroll
    for (int m = 0; m < 2; ++m) {
#pragma unroll
      for (int i = 0; i < 8; ++i) {
        int row = m * 16 + i + (lhalf << 3);
        if (row < nvalid) {
          float g = sGate[row];
          int t = sTok[row];
          atomicAdd(out + (size_t)t * DIMK + col, g * (acc2[j][m][i] + bv));
        }
      }
    }
  }
}

extern "C" void kernel_launch(void* const* d_in, const int* in_sizes, int n_in,
                              void* d_out, int out_size, void* d_ws, size_t ws_size,
                              hipStream_t stream) {
  const float* hidden = (const float*)d_in[0];
  const float* topw   = (const float*)d_in[1];
  const float* W12    = (const float*)d_in[2];
  const float* b12    = (const float*)d_in[3];
  const float* W3     = (const float*)d_in[4];
  const float* b3     = (const float*)d_in[5];
  const int*   topi   = (const int*)d_in[6];
  float* out = (float*)d_out;

  char* ws = (char*)d_ws;
  int*   counts     = (int*)(ws + 0);
  int*   cursor     = (int*)(ws + 32);
  int*   offsets    = (int*)(ws + 64);
  int*   slot_token = (int*)(ws + 128);
  float* slot_gate  = (float*)(ws + 128 + 4 * NSLOT);
  unsigned short* pw12h = (unsigned short*)(ws + 128 + 8 * NSLOT);  // 65664, 16B aligned
  unsigned short* pw12l = pw12h + (size_t)11534336;                 // 8*88*32*512
  unsigned short* pw3h  = pw12l + (size_t)11534336;
  unsigned short* pw3l  = pw3h + (size_t)5767168;                   // 8*64*22*512

  route_init<<<1, 32, 0, stream>>>(counts, cursor);
  route_count<<<NSLOT / 256, 256, 0, stream>>>(topi, counts);
  route_scan<<<1, 1, 0, stream>>>(counts, offsets, cursor);
  route_scatter<<<NSLOT / 256, 256, 0, stream>>>(topi, topw, cursor, slot_token, slot_gate);
  pack_w12<<<5632, 256, 0, stream>>>(W12, pw12h, pw12l);
  pack_w3<<<2816, 256, 0, stream>>>(W3, pw3h, pw3l);
  zero_f32<<<(T_TOK * DIMK / 4 + 255) / 256, 256, 0, stream>>>(out, T_TOK * DIMK / 4);

  dim3 grid(NSLOT / MROWS, E_EXP);
  moe_main<<<grid, 256, 0, stream>>>(hidden, b12, b3, pw12h, pw12l, pw3h, pw3l,
                                     offsets, slot_token, slot_gate, out);
}
